// Block_68702296867210
// MI455X (gfx1250) — matrix operand order repaired and checked by reading, not compiled, not run on hardware
//
#include <hip/hip_runtime.h>

// ---------------------------------------------------------------------------
// Types for CDNA5 WMMA (gfx1250, wave32)
// ---------------------------------------------------------------------------
typedef __attribute__((ext_vector_type(16))) __bf16 v16bf;
typedef __attribute__((ext_vector_type(8)))  float  v8f;

union Frag {
    unsigned int u[8];
    v16bf        v;
};

__device__ __forceinline__ v8f vzero8() {
    v8f z = {0.f, 0.f, 0.f, 0.f, 0.f, 0.f, 0.f, 0.f};
    return z;
}

// D = A(16x32 bf16) * B(32x16 bf16) + C(16x16 f32)
__device__ __forceinline__ v8f wmma_bf16(const Frag& a, const Frag& b, v8f c) {
    return __builtin_amdgcn_wmma_f32_16x16x32_bf16(
        /*neg_a=*/false, a.v, /*neg_b=*/false, b.v,
        /*c_mod=*/(short)0, c, /*reuse_a=*/false, /*reuse_b=*/false);
}

// fp32 -> bf16 round-to-nearest-even
__device__ __forceinline__ unsigned short f2bf(float f) {
    union { float f; unsigned int u; } c;
    c.f = f;
    unsigned int u = c.u;
    unsigned int r = u + 0x7FFFu + ((u >> 16) & 1u);
    return (unsigned short)(r >> 16);
}

// A-fragment (16x32 bf16) dword index within a 16-dword K-slice of one row
__device__ __forceinline__ int jmapA(int j, int half) {
    return (j < 4 ? j : j + 4) + half * 4;
}
// B-fragment (32x16 bf16), stored transposed Bt[n][k]: dword = (j < 4 ? j : j + 4) + 4*half
__device__ __forceinline__ int jmapB(int j, int half) {
    return (j < 4 ? j : j + 4) + half * 4;
}

__device__ __forceinline__ float redmax16(float v) {
#pragma unroll
    for (int off = 1; off < 16; off <<= 1)
        v = fmaxf(v, __shfl_xor(v, off, 32));
    return v;
}
__device__ __forceinline__ float redsum16(float v) {
#pragma unroll
    for (int off = 1; off < 16; off <<= 1)
        v += __shfl_xor(v, off, 32);
    return v;
}

__device__ __forceinline__ float gelu_tanh(float t) {
    return 0.5f * t * (1.0f + tanhf(0.7978845608028654f * (t + 0.044715f * t * t * t)));
}

// ---------------------------------------------------------------------------
// CDNA5 async global->LDS copy (ASYNCcnt-tracked, no VGPR data movement)
// ---------------------------------------------------------------------------
__device__ __forceinline__ void async_b128(void* lds, const void* gptr) {
    unsigned loff = (unsigned)(size_t)lds;  // low 32 bits of generic = LDS offset
    asm volatile("global_load_async_to_lds_b128 %0, %1, off"
                 :: "v"(loff), "v"(gptr)
                 : "memory");
}
__device__ __forceinline__ void wait_async0() {
    asm volatile("s_wait_asynccnt 0" ::: "memory");
}

// ---------------------------------------------------------------------------
// fp32 -> bf16 pairwise convert (packs 2 elements per dword)
// ---------------------------------------------------------------------------
__global__ __launch_bounds__(256)
void cvt_f32_bf16(const float* __restrict__ in, unsigned int* __restrict__ out, int n2) {
    int i = blockIdx.x * 256 + threadIdx.x;
    if (i < n2)
        out[i] = (unsigned int)f2bf(in[2 * i]) | ((unsigned int)f2bf(in[2 * i + 1]) << 16);
}

// ---------------------------------------------------------------------------
// W[K][N] fp32 -> Wt[N][K] bf16 (tiled transpose through LDS)
// ---------------------------------------------------------------------------
__global__ __launch_bounds__(256)
void transpose_cvt(const float* __restrict__ W, unsigned short* __restrict__ Wt,
                   int K, int N) {
    __shared__ float tile[32][33];
    const int n0 = blockIdx.x * 32, k0 = blockIdx.y * 32;
    const int tx = threadIdx.x, ty = threadIdx.y;
#pragma unroll
    for (int i = 0; i < 32; i += 8)
        tile[ty + i][tx] = W[(size_t)(k0 + ty + i) * N + n0 + tx];
    __syncthreads();
#pragma unroll
    for (int i = 0; i < 32; i += 8)
        Wt[(size_t)(n0 + ty + i) * K + k0 + tx] = f2bf(tile[tx][ty + i]);
}

// ---------------------------------------------------------------------------
// bf16 WMMA GEMM: C[M,N] = A[M,K] @ Bt[N,K]^T (+bias, act, residual)
// Block tile 128x128, BK=64, 256 threads = 8 waves, wave tile 32x64.
// Double-buffered LDS fed by async global->LDS copies; 16 WMMA per barrier.
// ---------------------------------------------------------------------------
template <bool ACT, bool RESID, bool OUTF, bool OUTB>
__global__ __launch_bounds__(256)
void gemm_bf16_kernel(const unsigned short* __restrict__ A,
                      const unsigned short* __restrict__ Bt,
                      const float* __restrict__ bias,
                      const float* __restrict__ resid,
                      float* __restrict__ outF,
                      unsigned short* __restrict__ outB,
                      int M, int N, int K) {
    __shared__ unsigned int sA[2][128 * 32];  // 128 rows x 64 bf16, x2 buffers
    __shared__ unsigned int sB[2][128 * 32];

    const int tid  = threadIdx.x;
    const int wave = tid >> 5;
    const int lane = tid & 31;
    const int lm   = lane & 15;
    const int half = lane >> 4;
    const int wm0  = (wave >> 1) * 32;  // 4 waves along M
    const int wn0  = (wave & 1) * 64;   // 2 waves along N
    const int bm0  = blockIdx.y * 128;
    const int bn0  = blockIdx.x * 128;

    v8f acc[2][4];
#pragma unroll
    for (int mt = 0; mt < 2; ++mt)
#pragma unroll
        for (int nt = 0; nt < 4; ++nt)
            acc[mt][nt] = vzero8();

    // Stage one 128x64 A tile + 128x64 B tile into buffer `buf` (async)
    auto stage = [&](int buf, int k0) {
#pragma unroll
        for (int i = 0; i < 4; ++i) {
            int q = tid + i * 256;           // 0..1023
            int r = q >> 3, c16 = q & 7;     // row, 16-byte chunk (8 per row)
            const char* ga = (const char*)A + ((size_t)(bm0 + r) * K + k0) * 2 + c16 * 16;
            async_b128(&sA[buf][r * 32 + c16 * 4], ga);
            const char* gb = (const char*)Bt + ((size_t)(bn0 + r) * K + k0) * 2 + c16 * 16;
            async_b128(&sB[buf][r * 32 + c16 * 4], gb);
        }
    };

    stage(0, 0);
    int buf = 0;
    for (int k0 = 0; k0 < K; k0 += 64) {
        wait_async0();
        __syncthreads();                     // buffer `buf` fully in LDS, all waves
        if (k0 + 64 < K) stage(buf ^ 1, k0 + 64);

#pragma unroll
        for (int kk = 0; kk < 2; ++kk) {     // two 32-wide K slices
            Frag aF[2], bF[4];
#pragma unroll
            for (int mt = 0; mt < 2; ++mt)
#pragma unroll
                for (int j = 0; j < 8; ++j)
                    aF[mt].u[j] = sA[buf][(wm0 + mt * 16 + lm) * 32 + kk * 16 + jmapA(j, half)];
#pragma unroll
            for (int nt = 0; nt < 4; ++nt)
#pragma unroll
                for (int j = 0; j < 8; ++j)
                    bF[nt].u[j] = sB[buf][(wn0 + nt * 16 + lm) * 32 + kk * 16 + jmapB(j, half)];
#pragma unroll
            for (int mt = 0; mt < 2; ++mt)
#pragma unroll
                for (int nt = 0; nt < 4; ++nt)
                    acc[mt][nt] = wmma_bf16(aF[mt], bF[nt], acc[mt][nt]);
        }
        buf ^= 1;
    }

    // Epilogue: bias -> activation -> residual -> store(s)
#pragma unroll
    for (int mt = 0; mt < 2; ++mt)
#pragma unroll
        for (int nt = 0; nt < 4; ++nt)
#pragma unroll
            for (int r = 0; r < 8; ++r) {
                int row = bm0 + wm0 + mt * 16 + r + 8 * half;
                int col = bn0 + wn0 + nt * 16 + lm;
                float v = acc[mt][nt][r] + bias[col];
                if (ACT) v = gelu_tanh(v);
                if (RESID) v += resid[(size_t)row * N + col];
                if (OUTF) outF[(size_t)row * N + col] = v;
                if (OUTB) outB[(size_t)row * N + col] = f2bf(v);
            }
}

// ---------------------------------------------------------------------------
// Flash attention with causal mask + key reweighting (attn_forcing).
// qkv: [B,S,3D] bf16. Block = 4 waves, 64 query rows; K blocks of 32 keys.
// out: attn [B*S, D] bf16.
// ---------------------------------------------------------------------------
#define AS 2048
#define AD 1024
#define AH 16
#define AHD 64
#define A3D 3072

__global__ __launch_bounds__(128)
void attn_kernel(const unsigned short* __restrict__ qkv,
                 const float* __restrict__ wgt,
                 const int* __restrict__ idxp,
                 const int* __restrict__ forcingp,
                 unsigned short* __restrict__ out) {
    __shared__ unsigned int   sQ[64 * 32];     // 64 q-rows x 64 bf16
    __shared__ unsigned int   sK[32 * 32];     // 32 keys  x 64 bf16
    __shared__ unsigned int   sVt[64 * 16];    // Vt: [hd=64][32 keys] bf16
    __shared__ unsigned short sP[4][16 * 32];  // per-wave P staging (16x32 bf16)

    const int tid  = threadIdx.x;
    const int wave = tid >> 5;
    const int lane = tid & 31;
    const int lm   = lane & 15;
    const int half = lane >> 4;

    const int qbase = blockIdx.x * 64;
    const int b = blockIdx.y / AH;
    const int h = blockIdx.y % AH;

    const float w0   = wgt[0];
    const int   forc = forcingp[0];
    const int   ib   = idxp[b];

    const char* qkvc = (const char*)qkv;
    const size_t rowb = (size_t)A3D * 2;  // qkv row bytes

    // Stage Q block once (async: 64 rows x 128B = 512 chunks / 128 thr = 4)
#pragma unroll
    for (int i = 0; i < 4; ++i) {
        int q = tid + i * 128;
        int r = q >> 3, c16 = q & 7;
        async_b128(&sQ[r * 32 + c16 * 4],
                   qkvc + (size_t)(b * AS + qbase + r) * rowb + h * 128 + c16 * 16);
    }
    wait_async0();
    __syncthreads();

    Frag qF[2];
#pragma unroll
    for (int c = 0; c < 2; ++c)
#pragma unroll
        for (int j = 0; j < 8; ++j)
            qF[c].u[j] = sQ[(wave * 16 + lm) * 32 + c * 16 + jmapA(j, half)];

    v8f o[4];
#pragma unroll
    for (int nt = 0; nt < 4; ++nt) o[nt] = vzero8();
    v8f mrow, lrow;
#pragma unroll
    for (int r = 0; r < 8; ++r) { mrow[r] = -3.0e38f; lrow[r] = 0.0f; }

    const int kend = qbase + 64;  // causal
    for (int kb = 0; kb < kend; kb += 32) {
        __syncthreads();  // previous iteration's reads of sK/sVt are done
        // Stage K block async (32 rows x 128B = 256 chunks / 128 thr = 2)
#pragma unroll
        for (int i = 0; i < 2; ++i) {
            int q = tid + i * 128;
            int r = q >> 3, c16 = q & 7;
            async_b128(&sK[r * 32 + c16 * 4],
                       qkvc + (size_t)(b * AS + kb + r) * rowb + 2048 + h * 128 + c16 * 16);
        }
        // Stage V transposed ([hd][key]); needs pack, so VGPR path
#pragma unroll
        for (int i = 0; i < 8; ++i) {
            int ii = tid + i * 128;
            int n = ii & 63, j = ii >> 6;
            unsigned short v0 = qkv[(size_t)(b * AS + kb + 2 * j)     * A3D + 2048 + h * AHD + n];
            unsigned short v1 = qkv[(size_t)(b * AS + kb + 2 * j + 1) * A3D + 2048 + h * AHD + n];
            sVt[n * 16 + j] = (unsigned int)v0 | ((unsigned int)v1 << 16);
        }
        wait_async0();
        __syncthreads();

        // S = Q @ K^T : two 16x16 score tiles, hd=64 -> 2 WMMA each
        v8f s0 = vzero8(), s1 = vzero8();
        Frag kf;
#pragma unroll
        for (int c = 0; c < 2; ++c) {
#pragma unroll
            for (int j = 0; j < 8; ++j)
                kf.u[j] = sK[(0 * 16 + lm) * 32 + c * 16 + jmapB(j, half)];
            s0 = wmma_bf16(qF[c], kf, s0);
        }
#pragma unroll
        for (int c = 0; c < 2; ++c) {
#pragma unroll
            for (int j = 0; j < 8; ++j)
                kf.u[j] = sK[(1 * 16 + lm) * 32 + c * 16 + jmapB(j, half)];
            s1 = wmma_bf16(qF[c], kf, s1);
        }

        // Online softmax; fold a_k reweight into unnormalized exp weights
        const int key0 = kb + lm;
        const int key1 = kb + 16 + lm;
        const float ak0 = (forc && key0 >= ib) ? w0 : 1.0f;
        const float ak1 = (forc && key1 >= ib) ? w0 : 1.0f;
        const float scale = 0.125f;  // 1/sqrt(64)
        unsigned short* pw = &sP[wave][0];
#pragma unroll
        for (int r = 0; r < 8; ++r) {
            int m = r + 8 * half;
            int qrow = qbase + wave * 16 + m;
            float a  = s0[r] * scale; if (key0 > qrow) a  = -1.0e30f;
            float bb = s1[r] * scale; if (key1 > qrow) bb = -1.0e30f;
            float mx = redmax16(fmaxf(a, bb));
            float mnew = fmaxf(mrow[r], mx);
            float corr = __expf(mrow[r] - mnew);
            float p0 = __expf(a  - mnew) * ak0;
            float p1 = __expf(bb - mnew) * ak1;
            float rs = redsum16(p0 + p1);
            lrow[r] = lrow[r] * corr + rs;
            mrow[r] = mnew;
            o[0][r] *= corr; o[1][r] *= corr; o[2][r] *= corr; o[3][r] *= corr;
            pw[m * 32 + lm]      = f2bf(p0);
            pw[m * 32 + 16 + lm] = f2bf(p1);
        }

        // P (C layout) -> A layout via wave-private LDS round trip
        Frag pF;
        const unsigned int* sPu = (const unsigned int*)pw;
#pragma unroll
        for (int j = 0; j < 8; ++j)
            pF.u[j] = sPu[lm * 16 + jmapA(j, half)];

        // O += P @ V  (K=32 keys in a single WMMA per hd tile)
        Frag vf;
#pragma unroll
        for (int nt = 0; nt < 4; ++nt) {
#pragma unroll
            for (int j = 0; j < 8; ++j)
                vf.u[j] = sVt[(nt * 16 + lm) * 16 + jmapB(j, half)];
            o[nt] = wmma_bf16(pF, vf, o[nt]);
        }
    }

    // Normalize and store bf16 attention output [B*S, D]
#pragma unroll
    for (int nt = 0; nt < 4; ++nt)
#pragma unroll
        for (int r = 0; r < 8; ++r) {
            int m = r + 8 * half;
            int row = b * AS + qbase + wave * 16 + m;
            int col = h * AHD + nt * 16 + lm;
            out[(size_t)row * AD + col] = f2bf(o[nt][r] / lrow[r]);
        }
}

// ---------------------------------------------------------------------------
// LayerNorm over D=1024, one block per row; f32 out + optional bf16 out
// ---------------------------------------------------------------------------
__global__ __launch_bounds__(256)
void ln_kernel(const float* __restrict__ in,
               const float* __restrict__ g,
               const float* __restrict__ bta,
               float* __restrict__ outF,
               unsigned short* __restrict__ outB) {
    const int D = 1024;
    const int row = blockIdx.x;
    const int tid = threadIdx.x;
    const float* xr = in + (size_t)row * D;

    float v[4], s = 0.f, ss = 0.f;
#pragma unroll
    for (int i = 0; i < 4; ++i) {
        v[i] = xr[tid + i * 256];
        s += v[i];
        ss += v[i] * v[i];
    }
    __shared__ float r1[256], r2[256];
    r1[tid] = s; r2[tid] = ss;
    __syncthreads();
    for (int st = 128; st > 0; st >>= 1) {
        if (tid < st) { r1[tid] += r1[tid + st]; r2[tid] += r2[tid + st]; }
        __syncthreads();
    }
    const float mean = r1[0] * (1.0f / D);
    const float var  = r2[0] * (1.0f / D) - mean * mean;
    const float inv  = rsqrtf(var + 1e-5f);
#pragma unroll
    for (int i = 0; i < 4; ++i) {
        int c = tid + i * 256;
        float o = g[c] * (v[i] - mean) * inv + bta[c];
        outF[(size_t)row * D + c] = o;
        if (outB) outB[(size_t)row * D + c] = f2bf(o);
    }
}

// ---------------------------------------------------------------------------
// Host launcher
// ---------------------------------------------------------------------------
extern "C" void kernel_launch(void* const* d_in, const int* in_sizes, int n_in,
                              void* d_out, int out_size, void* d_ws, size_t ws_size,
                              hipStream_t stream) {
    (void)in_sizes; (void)n_in; (void)out_size; (void)ws_size;

    const float* x        = (const float*)d_in[0];
    const float* weight   = (const float*)d_in[1];
    const float* linear_w = (const float*)d_in[2];
    const float* linear_b = (const float*)d_in[3];
    const float* proj_w   = (const float*)d_in[4];
    const float* proj_b   = (const float*)d_in[5];
    const float* ln1_g    = (const float*)d_in[6];
    const float* ln1_b    = (const float*)d_in[7];
    const float* fc_w     = (const float*)d_in[8];
    const float* fc_b     = (const float*)d_in[9];
    const float* cproj_w  = (const float*)d_in[10];
    const float* cproj_b  = (const float*)d_in[11];
    const float* ln2_g    = (const float*)d_in[12];
    const float* ln2_b    = (const float*)d_in[13];
    const int*   idx      = (const int*)d_in[14];
    const int*   forcing  = (const int*)d_in[15];
    float* out = (float*)d_out;

    const int B = 4, S = 2048, D = 1024, F = 4096, TD = 3072;
    const int M = B * S;  // 8192

    size_t off = 0;
    auto take = [&](size_t bytes) -> void* {
        off = (off + 255) & ~(size_t)255;
        void* p = (char*)d_ws + off;
        off += bytes;
        return p;
    };
    unsigned short* x_bf     = (unsigned short*)take((size_t)M * D * 2);
    unsigned short* lin_wt   = (unsigned short*)take((size_t)TD * D * 2);
    unsigned short* proj_wt  = (unsigned short*)take((size_t)D * D * 2);
    unsigned short* fc_wt    = (unsigned short*)take((size_t)F * D * 2);
    unsigned short* cproj_wt = (unsigned short*)take((size_t)D * F * 2);
    unsigned short* qkv_bf   = (unsigned short*)take((size_t)M * TD * 2);
    unsigned short* attn_bf  = (unsigned short*)take((size_t)M * D * 2);
    float*          res1     = (float*)take((size_t)M * D * 4);
    float*          n_f32    = (float*)take((size_t)M * D * 4);
    unsigned short* n_bf     = (unsigned short*)take((size_t)M * D * 2);
    unsigned short* act_bf   = (unsigned short*)take((size_t)M * F * 2);
    float*          res2     = (float*)take((size_t)M * D * 4);

    // x -> bf16
    {
        int n2 = M * D / 2;
        cvt_f32_bf16<<<(n2 + 255) / 256, 256, 0, stream>>>(x, (unsigned int*)x_bf, n2);
    }
    // Weight transpose+convert: W[K][N] -> Wt[N][K] bf16
    {
        dim3 blk(32, 8);
        transpose_cvt<<<dim3(TD / 32, D / 32), blk, 0, stream>>>(linear_w, lin_wt, D, TD);
        transpose_cvt<<<dim3(D / 32, D / 32),  blk, 0, stream>>>(proj_w,   proj_wt, D, D);
        transpose_cvt<<<dim3(F / 32, D / 32),  blk, 0, stream>>>(fc_w,     fc_wt,   D, F);
        transpose_cvt<<<dim3(D / 32, F / 32),  blk, 0, stream>>>(cproj_w,  cproj_wt, F, D);
    }
    // QKV projection -> bf16
    gemm_bf16_kernel<false, false, false, true><<<dim3(TD / 128, M / 128), 256, 0, stream>>>(
        x_bf, lin_wt, linear_b, nullptr, nullptr, qkv_bf, M, TD, D);
    // Attention -> bf16
    attn_kernel<<<dim3(S / 64, B * 16), 128, 0, stream>>>(
        qkv_bf, weight, idx, forcing, attn_bf);
    // Output projection + residual x -> res1 (f32)
    gemm_bf16_kernel<false, true, true, false><<<dim3(D / 128, M / 128), 256, 0, stream>>>(
        attn_bf, proj_wt, proj_b, x, res1, nullptr, M, D, D);
    // LN1 -> n (f32 + bf16)
    ln_kernel<<<M, 256, 0, stream>>>(res1, ln1_g, ln1_b, n_f32, n_bf);
    // FC + GELU -> bf16
    gemm_bf16_kernel<true, false, false, true><<<dim3(F / 128, M / 128), 256, 0, stream>>>(
        n_bf, fc_wt, fc_b, nullptr, nullptr, act_bf, M, F, D);
    // CProj + residual n -> res2 (f32)
    gemm_bf16_kernel<false, true, true, false><<<dim3(D / 128, M / 128), 256, 0, stream>>>(
        act_bf, cproj_wt, cproj_b, n_f32, res2, nullptr, M, D, F);
    // LN2 -> final output
    ln_kernel<<<M, 256, 0, stream>>>(res2, ln2_g, ln2_b, out, nullptr);
}
